// Transolver_38749194944558
// MI455X (gfx1250) — matrix operand order repaired and checked by reading, not compiled, not run on hardware
//
#include <hip/hip_runtime.h>
#include <hip/hip_bf16.h>
#include <math.h>

#define LNUM 4
#define BN 4
#define CH 256
#define OUTC 4
#define HH 128
#define WW 128
#define PW 130              // padded width/height for conv input
#define HEADS 8
#define DHD 32
#define GS 64
#define NPTS (HH*WW)
#define INNER (HEADS*DHD)
#define HID (4*CH)

typedef __attribute__((ext_vector_type(16))) __bf16 v16bf;
typedef __attribute__((ext_vector_type(8)))  __bf16 v8bf;
typedef __attribute__((ext_vector_type(8)))  float v8f;
typedef __attribute__((ext_vector_type(8)))  unsigned short us8;
typedef __attribute__((ext_vector_type(4)))  unsigned short us4;
typedef __attribute__((ext_vector_type(4)))  int v4i;

#if __has_builtin(__builtin_amdgcn_global_load_async_to_lds_b128) && \
    __has_builtin(__builtin_amdgcn_s_wait_asynccnt)
#define HAVE_ASYNC_LDS 1
#else
#define HAVE_ASYNC_LDS 0
#endif

__device__ __forceinline__ unsigned short f2bf(float f){
    unsigned int u = __builtin_bit_cast(unsigned int, f);
    u += 0x7FFFu + ((u >> 16) & 1u);
    return (unsigned short)(u >> 16);
}
__device__ __forceinline__ float bf2f(unsigned short s){
    unsigned int u = ((unsigned int)s) << 16;
    return __builtin_bit_cast(float, u);
}

// ---- async (direct-to-LDS) 16B copy with sync fallback ----
__device__ __forceinline__ void cp_async_b128(const unsigned short* g, unsigned short* l){
#if HAVE_ASYNC_LDS
    __builtin_amdgcn_global_load_async_to_lds_b128(
        (__attribute__((address_space(1))) v4i*)g,
        (__attribute__((address_space(3))) v4i*)l, 0, 0);
#else
    *(us8*)l = *(const us8*)g;
#endif
}
__device__ __forceinline__ void cp_wait_more(){
#if HAVE_ASYNC_LDS
    __builtin_amdgcn_s_wait_asynccnt(2);   // allow the 2 just-issued next-stage loads
#endif
}
__device__ __forceinline__ void cp_wait_all(){
#if HAVE_ASYNC_LDS
    __builtin_amdgcn_s_wait_asynccnt(0);
#endif
}

// ---- WMMA fragment loads (CDNA5 ISA 7.12.2, 16-bit, wave32) ----
__device__ __forceinline__ v16bf frag_concat(const unsigned short* p0,
                                             const unsigned short* p1){
    v8bf lo = __builtin_bit_cast(v8bf, *(const us8*)p0);
    v8bf hi = __builtin_bit_cast(v8bf, *(const us8*)p1);
    return __builtin_shufflevector(lo, hi, 0,1,2,3,4,5,6,7,8,9,10,11,12,13,14,15);
}
__device__ __forceinline__ v16bf frag_a(const unsigned short* lds, int m0,
                                        int stride, int lane){
    int m = m0 + (lane & 15);
    int k = (lane & 16) ? 8 : 0;
    const unsigned short* p = lds + m * stride + k;
    return frag_concat(p, p + 16);
}
__device__ __forceinline__ v16bf frag_b(const unsigned short* lds, int n0,
                                        int stride, int lane){
    int n = n0 + (lane & 15);
    int k = (lane & 16) ? 8 : 0;
    const unsigned short* p = lds + n * stride + k;
    return frag_concat(p, p + 16);
}
__device__ __forceinline__ v8f wmma_bf16(v16bf a, v16bf b, v8f c){
    return __builtin_amdgcn_wmma_f32_16x16x32_bf16(
        false, a, false, b, (short)0, c, false, false);
}
#define V8F_ZERO {0.f,0.f,0.f,0.f,0.f,0.f,0.f,0.f}

// ---------------- fp32 [mats,K,Nn] -> bf16 transposed [mats,Nn,K] ----------------
__global__ void cvt_transpose_kernel(const float* __restrict__ in,
                                     unsigned short* __restrict__ out,
                                     int K, int Nn, int total){
    int i = blockIdx.x * blockDim.x + threadIdx.x;
    int stride = gridDim.x * blockDim.x;
    int mk = K * Nn;
    for (; i < total; i += stride){
        int mat = i / mk, rem = i - mat * mk;
        int k = rem / Nn, n = rem - k * Nn;
        out[(size_t)mat * mk + (size_t)n * K + k] = f2bf(in[i]);
    }
}

// ---------------- LayerNorm (C=256), bf16 out, optional padded layout ----------------
__global__ void layernorm_bf16_kernel(const float* __restrict__ x,
                                      const float* __restrict__ w,
                                      const float* __restrict__ b,
                                      unsigned short* __restrict__ out,
                                      int padded){
    int row = blockIdx.x;
    int t = threadIdx.x;
    float v = x[(size_t)row * CH + t];
    __shared__ float red[256];
    red[t] = v; __syncthreads();
    for (int s = 128; s > 0; s >>= 1){ if (t < s) red[t] += red[t+s]; __syncthreads(); }
    float mean = red[0] / CH; __syncthreads();
    float d = v - mean;
    red[t] = d * d; __syncthreads();
    for (int s = 128; s > 0; s >>= 1){ if (t < s) red[t] += red[t+s]; __syncthreads(); }
    float var = red[0] / CH;
    float r = rsqrtf(var + 1e-5f);
    float y = d * r * w[t] + b[t];
    size_t o;
    if (padded){
        int n = row % NPTS, img = row / NPTS;
        int py = (n >> 7) + 1, px = (n & 127) + 1;
        o = ((size_t)img * PW * PW + (size_t)py * PW + px) * CH + t;
    } else {
        o = (size_t)row * CH + t;
    }
    out[o] = f2bf(y);
}

// ---------------- 3x3 conv as implicit GEMM, async double-buffered ----------------
// xpad: [B,130,130,C] bf16 (zero borders) ; wgtT: [9,INNER,C] bf16 ; out: [B,N,INNER]
__global__ __launch_bounds__(256)
void conv3x3_wmma_kernel(const unsigned short* __restrict__ xpad,
                         const unsigned short* __restrict__ wgtT,
                         const float* __restrict__ bias,
                         unsigned short* __restrict__ out){
    const int ASTR = 48, BSTR = 40, NS = 9 * (CH / 32);   // 72 stages
    __shared__ unsigned short As [2][64 * ASTR];
    __shared__ unsigned short BsT[2][64 * BSTR];
    int tid = threadIdx.x, lane = tid & 31, wv = tid >> 5;
    int wm = wv & 1, wn = wv >> 1;               // waves 2(M) x 4(N)
    int tilesPerImg = NPTS / 64;
    int bimg = blockIdx.x / tilesPerImg;
    int n0   = (blockIdx.x % tilesPerImg) * 64;
    int o0   = blockIdx.y * 64;
    v8f acc0 = V8F_ZERO, acc1 = V8F_ZERO;

    int r = tid >> 2, ch = (tid & 3) * 8;
    int n = n0 + r;
    int py = (n >> 7) + 1, px = (n & 127) + 1;
    const unsigned short* aBase =
        xpad + ((size_t)bimg * PW * PW + (size_t)py * PW + px) * CH + ch;
    const unsigned short* bBase = wgtT + (size_t)(o0 + r) * CH + ch;
    unsigned short* aDst[2] = { &As[0][r*ASTR + ch],  &As[1][r*ASTR + ch]  };
    unsigned short* bDst[2] = { &BsT[0][r*BSTR + ch], &BsT[1][r*BSTR + ch] };

    auto issue = [&](int s, int buf){
        int tap = s >> 3, k0 = (s & 7) << 5;
        int ty = tap / 3;
        int dy = ty - 1, dx = tap - ty * 3 - 1;
        cp_async_b128(aBase + (size_t)(dy * PW + dx) * CH + k0, aDst[buf]);
        cp_async_b128(bBase + (size_t)tap * (INNER * CH) + k0, bDst[buf]);
    };
    issue(0, 0);
    for (int s = 0; s < NS; ++s){
        int cur = s & 1;
        if (s + 1 < NS){ issue(s + 1, cur ^ 1); cp_wait_more(); }
        else           { cp_wait_all(); }
        __syncthreads();
        v16bf bf = frag_b(BsT[cur], wn * 16, BSTR, lane);
        v16bf a0 = frag_a(As[cur],  wm * 32,      ASTR, lane);
        v16bf a1 = frag_a(As[cur],  wm * 32 + 16, ASTR, lane);
        acc0 = wmma_bf16(a0, bf, acc0);
        acc1 = wmma_bf16(a1, bf, acc1);
        __syncthreads();
    }
    int nl = o0 + wn * 16 + (lane & 15);
    float bv = bias[nl];
    int moff = (lane & 16) ? 8 : 0;
#pragma unroll
    for (int v = 0; v < 8; ++v){
        int m0 = wm * 32 + v + moff;
        out[((size_t)bimg * NPTS + n0 + m0) * INNER + nl]      = f2bf(acc0[v] + bv);
        out[((size_t)bimg * NPTS + n0 + m0 + 16) * INNER + nl] = f2bf(acc1[v] + bv);
    }
}

// ---------------- slice logits + softmax -> sw bf16 ----------------
__global__ void slice_softmax_kernel(const unsigned short* __restrict__ xmid,
                                     const float* __restrict__ slw,   // [DH,G]
                                     const float* __restrict__ slb,   // [G]
                                     const float* __restrict__ temp,  // [HEADS]
                                     unsigned short* __restrict__ sw){// [B,HEADS,N,G]
    __shared__ float wlds[DHD * GS];
    __shared__ float blds[GS];
    __shared__ float xlds[8][DHD];
    int tid = threadIdx.x;
    for (int i = tid; i < DHD * GS; i += 256) wlds[i] = slw[i];
    if (tid < GS) blds[tid] = slb[tid];
    int lane = tid & 31, wv = tid >> 5;
    int row = blockIdx.x * 8 + wv;           // over B*HEADS*N
    int n = row % NPTS;
    int bh = row / NPTS;
    int h = bh % HEADS, b = bh / HEADS;
    xlds[wv][lane] = bf2f(xmid[((size_t)b * NPTS + n) * INNER + h * DHD + lane]);
    __syncthreads();
    float tp = temp[h]; tp = fminf(fmaxf(tp, 0.1f), 5.0f);
    float inv = 1.0f / tp;
    int g0 = lane, g1 = lane + 32;
    float l0 = blds[g0], l1 = blds[g1];
#pragma unroll
    for (int d = 0; d < DHD; ++d){
        float xv = xlds[wv][d];
        l0 += xv * wlds[d * GS + g0];
        l1 += xv * wlds[d * GS + g1];
    }
    l0 *= inv; l1 *= inv;
    float mx = fmaxf(l0, l1);
    for (int off = 16; off > 0; off >>= 1) mx = fmaxf(mx, __shfl_xor(mx, off, 32));
    float e0 = __expf(l0 - mx), e1 = __expf(l1 - mx);
    float s = e0 + e1;
    for (int off = 16; off > 0; off >>= 1) s += __shfl_xor(s, off, 32);
    float r = 1.0f / s;
    size_t base = ((size_t)bh * NPTS + n) * GS;
    sw[base + g0] = f2bf(e0 * r);
    sw[base + g1] = f2bf(e1 * r);
}

// ---------------- norm[b,h,g] = sum_n sw ----------------
__global__ void norm_sum_kernel(const unsigned short* __restrict__ sw,
                                float* __restrict__ norm){
    int bh = blockIdx.x;
    int t = threadIdx.x;
    int g = t & 63, part = t >> 6;
    const unsigned short* p = sw + (size_t)bh * NPTS * GS;
    float s = 0.f;
    for (int n = part; n < NPTS; n += 4) s += bf2f(p[(size_t)n * GS + g]);
    __shared__ float red[256];
    red[t] = s; __syncthreads();
    if (t < 64) norm[bh * GS + t] = red[t] + red[t+64] + red[t+128] + red[t+192];
}

// ---------------- st[b,h,g,d] = (sw^T @ fxmid) / (norm+eps) ----------------
__global__ __launch_bounds__(256)
void st_wmma_kernel(const unsigned short* __restrict__ sw,     // [B,HEADS,N,G]
                    const unsigned short* __restrict__ fxmid,  // [B,N,INNER]
                    const float* __restrict__ norm,            // [B,HEADS,G]
                    float* __restrict__ st){                   // [B,HEADS,G,DH]
    const int TSTR = 40;
    __shared__ unsigned short St[GS  * TSTR]; // [g=64][k=32] (transposed stage)
    __shared__ unsigned short Ft[DHD * TSTR]; // [d=32][k=32] (transposed stage)
    int bh = blockIdx.x;
    int h = bh % HEADS, b = bh / HEADS;
    int tid = threadIdx.x, lane = tid & 31, wv = tid >> 5;
    int mt = wv & 3, nt = wv >> 2;            // 4 g-tiles x 2 d-tiles
    v8f acc = V8F_ZERO;
    const unsigned short* swp = sw + (size_t)bh * NPTS * GS;
    const unsigned short* fxp = fxmid + (size_t)b * NPTS * INNER + h * DHD;
    for (int k0 = 0; k0 < NPTS; k0 += 32){
        { int r = tid >> 3, c = (tid & 7) * 8;     // contiguous global, scatter LDS
          us8 v = *(const us8*)(swp + (size_t)(k0 + r) * GS + c);
#pragma unroll
          for (int j = 0; j < 8; ++j) St[(c + j) * TSTR + r] = v[j]; }
        { int r = tid >> 3, c = (tid & 7) * 4;
          us4 v = *(const us4*)(fxp + (size_t)(k0 + r) * INNER + c);
#pragma unroll
          for (int j = 0; j < 4; ++j) Ft[(c + j) * TSTR + r] = v[j]; }
        __syncthreads();
        v16bf a  = frag_a(St, mt * 16, TSTR, lane);
        v16bf bm = frag_b(Ft, nt * 16, TSTR, lane);
        acc = wmma_bf16(a, bm, acc);
        __syncthreads();
    }
    int d = nt * 16 + (lane & 15);
    int moff = (lane & 16) ? 8 : 0;
#pragma unroll
    for (int v = 0; v < 8; ++v){
        int g = mt * 16 + v + moff;
        float nv = norm[bh * GS + g] + 1e-5f;
        st[((size_t)bh * GS + g) * DHD + d] = acc[v] / nv;
    }
}

// ---------------- tiny slice-space attention (per b,h) ----------------
__global__ void slice_attn_kernel(const float* __restrict__ st,
                                  const float* __restrict__ wq,
                                  const float* __restrict__ wk,
                                  const float* __restrict__ wv_,
                                  float* __restrict__ ot){
    __shared__ float sst[GS][DHD];
    __shared__ float q[GS][DHD], kk[GS][DHD], vv[GS][DHD];
    __shared__ float wql[DHD*DHD], wkl[DHD*DHD], wvl[DHD*DHD];
    __shared__ float att[GS][GS];
    int bh = blockIdx.x;
    int t = threadIdx.x;
    const float* stp = st + (size_t)bh * GS * DHD;
    for (int i = t; i < GS * DHD; i += 256) ((float*)sst)[i] = stp[i];
    for (int i = t; i < DHD * DHD; i += 256){ wql[i]=wq[i]; wkl[i]=wk[i]; wvl[i]=wv_[i]; }
    __syncthreads();
    for (int i = t; i < GS * DHD; i += 256){
        int g = i / DHD, d = i % DHD;
        float aq = 0, ak = 0, av = 0;
        for (int c = 0; c < DHD; ++c){
            float s = sst[g][c];
            aq += s * wql[c*DHD + d]; ak += s * wkl[c*DHD + d]; av += s * wvl[c*DHD + d];
        }
        q[g][d] = aq; kk[g][d] = ak; vv[g][d] = av;
    }
    __syncthreads();
    const float scale = 0.1767766952966369f; // DH^-0.5
    for (int i = t; i < GS * GS; i += 256){
        int g = i / GS, k2 = i % GS;
        float s = 0;
        for (int d = 0; d < DHD; ++d) s += q[g][d] * kk[k2][d];
        att[g][k2] = s * scale;
    }
    __syncthreads();
    if (t < GS){
        float mx = -1e30f;
        for (int k2 = 0; k2 < GS; ++k2) mx = fmaxf(mx, att[t][k2]);
        float s = 0;
        for (int k2 = 0; k2 < GS; ++k2){ float e = __expf(att[t][k2] - mx); att[t][k2] = e; s += e; }
        float r = 1.f / s;
        for (int k2 = 0; k2 < GS; ++k2) att[t][k2] *= r;
    }
    __syncthreads();
    float* otp = ot + (size_t)bh * GS * DHD;
    for (int i = t; i < GS * DHD; i += 256){
        int g = i / DHD, d = i % DHD;
        float s = 0;
        for (int k2 = 0; k2 < GS; ++k2) s += att[g][k2] * vv[k2][d];
        otp[i] = s;
    }
}

// ---------------- deslice: out[b,n,h*DH+d] = sw[n,:] @ ot[:,d] ----------------
__global__ __launch_bounds__(256)
void deslice_wmma_kernel(const unsigned short* __restrict__ sw, // [B,HEADS,N,G]
                         const float* __restrict__ ot,          // [B,HEADS,G,DH]
                         unsigned short* __restrict__ outm){    // [B,N,INNER]
    const int ASTR = 72, BSTR = 72;
    __shared__ unsigned short As [128 * ASTR]; // [m=128][k=64]
    __shared__ unsigned short BsT[DHD * BSTR]; // [n=32][k=64] (ot transposed)
    int bh = blockIdx.y;
    int h = bh % HEADS, b = bh / HEADS;
    int n0 = blockIdx.x * 128;
    int tid = threadIdx.x, lane = tid & 31, wv = tid >> 5;
    const float* otp = ot + (size_t)bh * GS * DHD;
    for (int i = tid; i < GS * DHD; i += 256){
        int g = i & 63, d = i >> 6;
        BsT[d * BSTR + g] = f2bf(otp[(size_t)g * DHD + d]);
    }
    const unsigned short* swp = sw + ((size_t)bh * NPTS + n0) * GS;
    for (int i = tid; i < 128 * 8; i += 256){
        int r = i >> 3, c = (i & 7) * 8;
        *(us8*)(As + r * ASTR + c) = *(const us8*)(swp + (size_t)r * GS + c);
    }
    __syncthreads();
    v8f acc0 = V8F_ZERO, acc1 = V8F_ZERO;
#pragma unroll
    for (int ks = 0; ks < 2; ++ks){
        v16bf a  = frag_a(As  + ks * 32, wv * 16, ASTR, lane);
        v16bf b0 = frag_b(BsT + ks * 32, 0,  BSTR, lane);
        v16bf b1 = frag_b(BsT + ks * 32, 16, BSTR, lane);
        acc0 = wmma_bf16(a, b0, acc0);
        acc1 = wmma_bf16(a, b1, acc1);
    }
    unsigned short* op = outm + ((size_t)b * NPTS + n0) * INNER + h * DHD;
    int nl = lane & 15;
    int moff = (lane & 16) ? 8 : 0;
#pragma unroll
    for (int v = 0; v < 8; ++v){
        int m = wv * 16 + v + moff;
        op[(size_t)m * INNER + nl]      = f2bf(acc0[v]);
        op[(size_t)m * INNER + 16 + nl] = f2bf(acc1[v]);
    }
}

// ---------------- generic WMMA GEMM, async double-buffered ----------------
// A: [M,K] bf16 row-major ; BmT: [Nn,K] bf16 (pre-transposed weights)
template<bool GELU, bool OUT_BF16>
__global__ __launch_bounds__(256)
void gemm_wmma_kernel(const unsigned short* __restrict__ A,
                      const unsigned short* __restrict__ BmT,
                      const float* __restrict__ bias,
                      const float* __restrict__ resid,
                      float* __restrict__ outF,
                      unsigned short* __restrict__ outB,
                      int K, int Nn){
    const int ASTR = 48, BSTR = 40;
    __shared__ unsigned short As [2][64 * ASTR];
    __shared__ unsigned short BsT[2][64 * BSTR];
    int tid = threadIdx.x, lane = tid & 31, wv = tid >> 5;
    int wm = wv & 1, wn = wv >> 1;
    int m0 = blockIdx.x * 64;
    int o0 = blockIdx.y * 64;
    v8f acc0 = V8F_ZERO, acc1 = V8F_ZERO;
    int ra = tid >> 2, ca = (tid & 3) * 8;
    const unsigned short* ga = A   + (size_t)(m0 + ra) * K + ca;
    const unsigned short* gb = BmT + (size_t)(o0 + ra) * K + ca;
    unsigned short* aDst[2] = { &As[0][ra*ASTR + ca],  &As[1][ra*ASTR + ca]  };
    unsigned short* bDst[2] = { &BsT[0][ra*BSTR + ca], &BsT[1][ra*BSTR + ca] };
    int nk = K / 32;
    auto issue = [&](int ki, int buf){
        cp_async_b128(ga + ki * 32, aDst[buf]);
        cp_async_b128(gb + ki * 32, bDst[buf]);
    };
    issue(0, 0);
    for (int ki = 0; ki < nk; ++ki){
        int cur = ki & 1;
        if (ki + 1 < nk){ issue(ki + 1, cur ^ 1); cp_wait_more(); }
        else            { cp_wait_all(); }
        __syncthreads();
        v16bf bf = frag_b(BsT[cur], wn * 16, BSTR, lane);
        v16bf a0 = frag_a(As[cur],  wm * 32,      ASTR, lane);
        v16bf a1 = frag_a(As[cur],  wm * 32 + 16, ASTR, lane);
        acc0 = wmma_bf16(a0, bf, acc0);
        acc1 = wmma_bf16(a1, bf, acc1);
        __syncthreads();
    }
    int nl = o0 + wn * 16 + (lane & 15);
    float bv = bias ? bias[nl] : 0.f;
    int moff = (lane & 16) ? 8 : 0;
#pragma unroll
    for (int v = 0; v < 8; ++v){
#pragma unroll
        for (int half = 0; half < 2; ++half){
            int m = m0 + wm * 32 + half * 16 + v + moff;
            float val = (half ? acc1[v] : acc0[v]) + bv;
            if (GELU) val = 0.5f * val * (1.0f + erff(val * 0.7071067811865475f));
            if (resid) val += resid[(size_t)m * Nn + nl];
            if (OUT_BF16) outB[(size_t)m * Nn + nl] = f2bf(val);
            else          outF[(size_t)m * Nn + nl] = val;
        }
    }
}

// ---------------- final LN + linear head (C -> 4) ----------------
__global__ void head_kernel(const float* __restrict__ fx,
                            const float* __restrict__ ln3w, const float* __restrict__ ln3b,
                            const float* __restrict__ wout, const float* __restrict__ bout,
                            float* __restrict__ out){
    int row = blockIdx.x;
    int t = threadIdx.x;
    float v = fx[(size_t)row * CH + t];
    __shared__ float red[256];
    red[t] = v; __syncthreads();
    for (int s = 128; s > 0; s >>= 1){ if (t < s) red[t] += red[t+s]; __syncthreads(); }
    float mean = red[0] / CH; __syncthreads();
    float d = v - mean;
    red[t] = d * d; __syncthreads();
    for (int s = 128; s > 0; s >>= 1){ if (t < s) red[t] += red[t+s]; __syncthreads(); }
    float var = red[0] / CH; __syncthreads();
    float y = d * rsqrtf(var + 1e-5f) * ln3w[t] + ln3b[t];
#pragma unroll
    for (int j = 0; j < OUTC; ++j){
        red[t] = y * wout[t * OUTC + j]; __syncthreads();
        for (int s = 128; s > 0; s >>= 1){ if (t < s) red[t] += red[t+s]; __syncthreads(); }
        if (t == 0) out[(size_t)row * OUTC + j] = red[0] + bout[j];
        __syncthreads();
    }
}

extern "C" void kernel_launch(void* const* d_in, const int* in_sizes, int n_in,
                              void* d_out, int out_size, void* d_ws, size_t ws_size,
                              hipStream_t stream) {
    (void)in_sizes; (void)n_in; (void)out_size; (void)ws_size;
    const float* fx       = (const float*)d_in[0];
    const float* ln1_w    = (const float*)d_in[1];
    const float* ln1_b    = (const float*)d_in[2];
    const float* convx_w  = (const float*)d_in[3];
    const float* convx_b  = (const float*)d_in[4];
    const float* convfx_w = (const float*)d_in[5];
    const float* convfx_b = (const float*)d_in[6];
    const float* slice_w  = (const float*)d_in[7];
    const float* slice_b  = (const float*)d_in[8];
    const float* temperat = (const float*)d_in[9];
    const float* wq       = (const float*)d_in[10];
    const float* wk       = (const float*)d_in[11];
    const float* wv       = (const float*)d_in[12];
    const float* wo       = (const float*)d_in[13];
    const float* bo       = (const float*)d_in[14];
    const float* ln2_w    = (const float*)d_in[15];
    const float* ln2_b    = (const float*)d_in[16];
    const float* w1       = (const float*)d_in[17];
    const float* b1       = (const float*)d_in[18];
    const float* w2       = (const float*)d_in[19];
    const float* b2       = (const float*)d_in[20];
    const float* ln3_w    = (const float*)d_in[21];
    const float* ln3_b    = (const float*)d_in[22];
    const float* w_out    = (const float*)d_in[23];
    const float* b_out    = (const float*)d_in[24];

    const size_t M = (size_t)BN * NPTS;          // 65536 rows
    const size_t SZ_FX   = M * CH * 4;
    const size_t SZ_XLN  = M * CH * 2;
    const size_t SZ_XPAD = (size_t)BN * PW * PW * CH * 2;
    const size_t SZ_XMID = M * INNER * 2;
    const size_t SZ_HB   = M * HID * 2;          // aliased over xmid/fxmid/sw

    char* ws = (char*)d_ws;
    size_t off = 0;
    float*          fxbuf  = (float*)(ws + off);          off += SZ_FX;
    unsigned short* xln    = (unsigned short*)(ws + off); off += SZ_XLN;
    unsigned short* xpad   = (unsigned short*)(ws + off); off += SZ_XPAD;
    size_t offR = off;
    unsigned short* xmid   = (unsigned short*)(ws + offR);
    unsigned short* fxmid  = (unsigned short*)(ws + offR + SZ_XMID);
    unsigned short* swb    = (unsigned short*)(ws + offR + 2*SZ_XMID);
    unsigned short* hbuf   = (unsigned short*)(ws + offR);
    off = offR + SZ_HB;
    float*          normb  = (float*)(ws + off);          off += (size_t)BN*HEADS*GS*4;
    float*          stbuf  = (float*)(ws + off);          off += (size_t)BN*HEADS*GS*DHD*4;
    float*          otbuf  = (float*)(ws + off);          off += (size_t)BN*HEADS*GS*DHD*4;
    unsigned short* cvx    = (unsigned short*)(ws + off); off += (size_t)LNUM*9*CH*INNER*2;
    unsigned short* cvf    = (unsigned short*)(ws + off); off += (size_t)LNUM*9*CH*INNER*2;
    unsigned short* wob    = (unsigned short*)(ws + off); off += (size_t)LNUM*INNER*CH*2;
    unsigned short* w1b    = (unsigned short*)(ws + off); off += (size_t)LNUM*CH*HID*2;
    unsigned short* w2b    = (unsigned short*)(ws + off); off += (size_t)LNUM*HID*CH*2;

    (void)hipMemcpyAsync(fxbuf, fx, SZ_FX, hipMemcpyDeviceToDevice, stream);
    (void)hipMemsetAsync(xpad, 0, SZ_XPAD, stream);   // zero borders once per call

    cvt_transpose_kernel<<<4096, 256, 0, stream>>>(convx_w,  cvx, CH, INNER, LNUM*9*CH*INNER);
    cvt_transpose_kernel<<<4096, 256, 0, stream>>>(convfx_w, cvf, CH, INNER, LNUM*9*CH*INNER);
    cvt_transpose_kernel<<<1024, 256, 0, stream>>>(wo, wob, INNER, CH, LNUM*INNER*CH);
    cvt_transpose_kernel<<<4096, 256, 0, stream>>>(w1, w1b, CH, HID, LNUM*CH*HID);
    cvt_transpose_kernel<<<4096, 256, 0, stream>>>(w2, w2b, HID, CH, LNUM*HID*CH);

    const int nBlocks = (int)M;                  // 65536
    for (int i = 0; i < LNUM; ++i){
        layernorm_bf16_kernel<<<nBlocks, 256, 0, stream>>>(
            fxbuf, ln1_w + i*CH, ln1_b + i*CH, xpad, 1);
        conv3x3_wmma_kernel<<<dim3(BN*NPTS/64, INNER/64), 256, 0, stream>>>(
            xpad, cvf + (size_t)i*9*CH*INNER, convfx_b + i*INNER, fxmid);
        conv3x3_wmma_kernel<<<dim3(BN*NPTS/64, INNER/64), 256, 0, stream>>>(
            xpad, cvx + (size_t)i*9*CH*INNER, convx_b + i*INNER, xmid);
        slice_softmax_kernel<<<BN*HEADS*NPTS/8, 256, 0, stream>>>(
            xmid, slice_w + i*DHD*GS, slice_b + i*GS, temperat + i*HEADS, swb);
        norm_sum_kernel<<<BN*HEADS, 256, 0, stream>>>(swb, normb);
        st_wmma_kernel<<<BN*HEADS, 256, 0, stream>>>(swb, fxmid, normb, stbuf);
        slice_attn_kernel<<<BN*HEADS, 256, 0, stream>>>(
            stbuf, wq + i*DHD*DHD, wk + i*DHD*DHD, wv + i*DHD*DHD, otbuf);
        deslice_wmma_kernel<<<dim3(NPTS/128, BN*HEADS), 256, 0, stream>>>(swb, otbuf, xln);
        gemm_wmma_kernel<false, false><<<dim3(nBlocks/64, CH/64), 256, 0, stream>>>(
            xln, wob + (size_t)i*INNER*CH, bo + i*CH, fxbuf, fxbuf, nullptr, INNER, CH);
        layernorm_bf16_kernel<<<nBlocks, 256, 0, stream>>>(
            fxbuf, ln2_w + i*CH, ln2_b + i*CH, xln, 0);
        gemm_wmma_kernel<true, true><<<dim3(nBlocks/64, HID/64), 256, 0, stream>>>(
            xln, w1b + (size_t)i*CH*HID, b1 + i*HID, nullptr, nullptr, hbuf, CH, HID);
        gemm_wmma_kernel<false, false><<<dim3(nBlocks/64, CH/64), 256, 0, stream>>>(
            hbuf, w2b + (size_t)i*HID*CH, b2 + i*CH, fxbuf, fxbuf, nullptr, HID, CH);
    }
    head_kernel<<<nBlocks, 256, 0, stream>>>(fxbuf, ln3_w, ln3_b, w_out, b_out, (float*)d_out);
}
